// LowRank_Attention_47158740910458
// MI455X (gfx1250) — matrix operand order repaired and checked by reading, not compiled, not run on hardware
//
#include <hip/hip_runtime.h>
#include <hip/hip_bf16.h>

typedef __attribute__((ext_vector_type(16))) _Float16 v16h;
typedef __attribute__((ext_vector_type(8)))  float    v8f;

#define HID   768
#define SEQ   197
#define SPAD  224        // 197 padded to 7*32
#define NTOK  (64*197)   // 12608, divisible by 16
#define MTILES 788

union F16Frag { v16h v; uint4 u[2]; };

// A-fragment: 16-bit A 16x32 layout. lane<16: k in {0..7}∪{16..23}; lane>=16: +8.
//   p0 = row_base + (lane>>4)*8 + kk ; p1 = p0 + 16
// B-fragment: 32x16 column-per-lane, same k order as A.
//   p0 = col_base + (lane>>4)*8 + kk ; p1 = p0 + 16
static __device__ __forceinline__ v16h load_frag(const _Float16* p0, const _Float16* p1) {
    F16Frag f;
    f.u[0] = *(const uint4*)p0;
    f.u[1] = *(const uint4*)p1;
    return f.v;
}

static __device__ __forceinline__ v8f wmma_f16(v16h a, v16h b, v8f c) {
    return __builtin_amdgcn_wmma_f32_16x16x32_f16(false, a, false, b, (short)0, c, false, false);
}

// ---------------- helper kernels ----------------

// 8 elements per thread: 2x b128 load -> 1x b128 store
__global__ void cvt_f32_to_f16_v8(const float* __restrict__ x, _Float16* __restrict__ y, int n8) {
    int i = blockIdx.x * 256 + threadIdx.x;
    if (i >= n8) return;
    const float4* px = (const float4*)x + (size_t)i * 2;
    float4 a = px[0], b = px[1];
    union { _Float16 h[8]; uint4 u; } o;
    o.h[0] = (_Float16)a.x; o.h[1] = (_Float16)a.y; o.h[2] = (_Float16)a.z; o.h[3] = (_Float16)a.w;
    o.h[4] = (_Float16)b.x; o.h[5] = (_Float16)b.y; o.h[6] = (_Float16)b.z; o.h[7] = (_Float16)b.w;
    ((uint4*)y)[i] = o.u;
}

// WT[n][k] = (f16) W[k][n]   (W is [768 in][768 out])
__global__ void transpose_w(const float* __restrict__ W, _Float16* __restrict__ WT) {
    __shared__ _Float16 tile[64][66];
    int nb = (blockIdx.x % 12) * 64;
    int kb = (blockIdx.x / 12) * 64;
    int c  = threadIdx.x & 63;
    int r0 = threadIdx.x >> 6;   // 0..3
    #pragma unroll
    for (int r = r0; r < 64; r += 4)
        tile[r][c] = (_Float16)W[(size_t)(kb + r) * HID + (nb + c)];
    __syncthreads();
    #pragma unroll
    for (int r = r0; r < 64; r += 4)
        WT[(size_t)(nb + r) * HID + (kb + c)] = tile[c][r];
}

// ET[r][s] = (f16) E[s][r], zero-padded s in [197,224)
__global__ void transpose_e(const float* __restrict__ E, _Float16* __restrict__ ET) {
    int i = blockIdx.x * 256 + threadIdx.x;
    if (i >= 64 * SPAD) return;
    int r = i / SPAD, s = i % SPAD;
    ET[i] = (s < SEQ) ? (_Float16)E[s * 64 + r] : (_Float16)0.f;
}

// ---------------- main GEMM: C[M,768] = A[M,768] @ W + bias ----------------
// WT is [768 out][768 in] row-major f16.
// Block = 8 waves, tile 256(M) x 64(N). Each wave: 2 M-subtiles (32M x 64N).
// B tile (64 x 32k) double-buffered in LDS, shared by all waves.

#define LDB 40   // padded LDS row stride (halves): 80B -> bank-conflict-free fragment reads

template <bool HALF_OUT>
__global__ void __launch_bounds__(256) gemm_768(const _Float16* __restrict__ A,
                                                const _Float16* __restrict__ WT,
                                                const float* __restrict__ bias,
                                                void* __restrict__ out) {
    __shared__ _Float16 LB[2][64 * LDB];

    int tid = threadIdx.x, wid = tid >> 5, lane = tid & 31;
    int lo = lane & 15, hi = lane >> 4;
    int mb = blockIdx.x / 12, ng = blockIdx.x % 12;
    int n0 = ng * 64;
    int mt0 = mb * 16 + wid * 2;                // wave's first m-tile
    bool run0 = (mt0     < MTILES);
    bool run1 = (mt0 + 1 < MTILES);

    const _Float16* arow0 = A + (size_t)(mt0 * 16 + lo) * HID + (hi << 3);
    const _Float16* arow1 = arow0 + (size_t)16 * HID;

    // staging map: thread t -> B row n=t>>2, 16B segment seg=t&3
    int sn = tid >> 2, sseg = tid & 3;
    const _Float16* gsrc = WT + (size_t)(n0 + sn) * HID + sseg * 8;
    _Float16* sdst[2] = { &LB[0][sn * LDB + sseg * 8], &LB[1][sn * LDB + sseg * 8] };

    uint4 r = *(const uint4*)gsrc;
    *(uint4*)sdst[0] = r;

    v8f acc[8];
    #pragma unroll
    for (int i = 0; i < 8; ++i) acc[i] = (v8f){};

    for (int kt = 0; kt < 24; ++kt) {
        int cur = kt & 1;
        if (kt < 23) r = *(const uint4*)(gsrc + (kt + 1) * 32);
        __syncthreads();                         // LB[cur] ready; prior compute done
        if (kt < 23) *(uint4*)sdst[cur ^ 1] = r; // fill next buffer

        const _Float16* LBc = &LB[cur][0];
        int kkA = kt * 32;
        v16h a0 = {}, a1 = {};
        if (run0) a0 = load_frag(arow0 + kkA, arow0 + kkA + 16);
        if (run1) a1 = load_frag(arow1 + kkA, arow1 + kkA + 16);
        #pragma unroll
        for (int nt = 0; nt < 4; ++nt) {
            const _Float16* bp = LBc + (nt * 16 + lo) * LDB + (hi << 3);
            v16h bb = load_frag(bp, bp + 16);
            if (run0) acc[nt]     = wmma_f16(a0, bb, acc[nt]);
            if (run1) acc[4 + nt] = wmma_f16(a1, bb, acc[4 + nt]);
        }
    }

    #pragma unroll
    for (int sub = 0; sub < 2; ++sub) {
        if (sub == 0 ? !run0 : !run1) continue;
        int m0 = (mt0 + sub) * 16;
        #pragma unroll
        for (int nt = 0; nt < 4; ++nt) {
            int col = n0 + nt * 16 + lo;
            float bv = bias[col];
            v8f a = acc[sub * 4 + nt];
            #pragma unroll
            for (int i = 0; i < 8; ++i) {
                int row = m0 + i + (hi << 3);
                float vv = a[i] + bv;
                if (HALF_OUT) ((_Float16*)out)[(size_t)row * HID + col] = (_Float16)vv;
                else          ((float*)out)[(size_t)row * HID + col]    = vv;
            }
        }
    }
}

// ---------------- fused low-rank attention core (one block per b,h) ----------------
// pk = Ek^T K, pv = Ev^T V ; scores = Q pk^T * scale ; softmax ; ctx = P pv

__global__ void lr_attn_core(const _Float16* __restrict__ Q, const _Float16* __restrict__ K,
                             const _Float16* __restrict__ V, const _Float16* __restrict__ EkT,
                             const _Float16* __restrict__ EvT, _Float16* __restrict__ CTX) {
    __shared__ char smem[100352];
    _Float16* KT  = (_Float16*)(smem);            // [64][224]  (overlapped by SCF later)
    _Float16* VT  = (_Float16*)(smem + 28672);    // [64][224]
    float*    SCF = (float*)(smem);               // [208][64] f32, overlaps KT/VT
    _Float16* PK  = (_Float16*)(smem + 57344);    // [64][64]  pk[r][d]
    _Float16* PVT = (_Float16*)(smem + 65536);    // [64][64]  pv^T[d][r]
    _Float16* PRB = (_Float16*)(smem + 73728);    // [208][64] probs f16

    int bh = blockIdx.x;
    int b = bh / 12, h = bh % 12;
    int tid = threadIdx.x, wid = tid >> 5, lane = tid & 31;
    int lo = lane & 15, hi = lane >> 4;
    size_t base = (size_t)b * SEQ * HID + h * 64;

    // phase 1: load K,V head slice transposed into LDS (zero-padded seq)
    for (int i = tid; i < 64 * SPAD; i += 256) {
        int d = i / SPAD, s = i % SPAD;
        _Float16 kv = (_Float16)0.f, vv = (_Float16)0.f;
        if (s < SEQ) {
            kv = K[base + (size_t)s * HID + d];
            vv = V[base + (size_t)s * HID + d];
        }
        KT[i] = kv; VT[i] = vv;
    }
    __syncthreads();

    // phase 2: pk[r][d] and pv^T[d][r], 16+16 tiles of 16x16, K=224
    for (int job = wid; job < 32; job += 8) {
        bool isV = job >= 16;
        int t  = job & 15;
        int r0 = (t >> 2) * 16, d0 = (t & 3) * 16;
        const _Float16* E = isV ? EvT : EkT;
        const _Float16* T = isV ? VT  : KT;
        const _Float16* ap = E + (r0 + lo) * SPAD + (hi << 3);
        const _Float16* bp = T + (d0 + lo) * SPAD + (hi << 3);
        v8f acc = {};
        #pragma unroll
        for (int kk = 0; kk < SPAD; kk += 32) {
            v16h a = load_frag(ap + kk, ap + kk + 16);
            v16h bb = load_frag(bp + kk, bp + kk + 16);
            acc = wmma_f16(a, bb, acc);
        }
        #pragma unroll
        for (int i = 0; i < 8; ++i) {
            int rr = r0 + i + (hi << 3);
            int dd = d0 + lo;
            if (isV) PVT[dd * 64 + rr] = (_Float16)acc[i];
            else     PK[rr * 64 + dd]  = (_Float16)acc[i];
        }
    }
    __syncthreads();

    // phase 3: scores[s][r] = scale * Q[s,:] . pk[r,:]  (13 x 4 tiles, K=64)
    const float scale = 0.125f;   // 1/sqrt(64)
    for (int job = wid; job < 52; job += 8) {
        int s0 = (job >> 2) * 16, r0 = (job & 3) * 16;
        int s = s0 + lo; if (s > SEQ - 1) s = SEQ - 1;   // clamp pad rows (discarded below)
        const _Float16* ap = Q + base + (size_t)s * HID + (hi << 3);
        const _Float16* bp = PK + (r0 + lo) * 64 + (hi << 3);
        v8f acc = {};
        #pragma unroll
        for (int kk = 0; kk < 64; kk += 32) {
            v16h a = load_frag(ap + kk, ap + kk + 16);
            v16h bb = load_frag(bp + kk, bp + kk + 16);
            acc = wmma_f16(a, bb, acc);
        }
        #pragma unroll
        for (int i = 0; i < 8; ++i)
            SCF[(s0 + i + (hi << 3)) * 64 + r0 + lo] = acc[i] * scale;
    }
    __syncthreads();

    // phase 4: row softmax over rank axis (64), f32 in LDS, probs -> f16
    for (int t = tid; t < 208; t += 256) {
        if (t < SEQ) {
            float m = -1e30f;
            for (int j = 0; j < 64; ++j) { float x = SCF[t * 64 + j]; m = x > m ? x : m; }
            float sum = 0.f;
            for (int j = 0; j < 64; ++j) {
                float e = __expf(SCF[t * 64 + j] - m);
                SCF[t * 64 + j] = e;   // thread-exclusive row, in place
                sum += e;
            }
            float inv = 1.f / sum;
            for (int j = 0; j < 64; ++j)
                PRB[t * 64 + j] = (_Float16)(SCF[t * 64 + j] * inv);
        } else {
            for (int j = 0; j < 64; ++j) PRB[t * 64 + j] = (_Float16)0.f;
        }
    }
    __syncthreads();

    // phase 5: ctx[s][d] = P[s,:] . pv[:,d]  (B from pv^T rows), K=64
    for (int job = wid; job < 52; job += 8) {
        int s0 = (job >> 2) * 16, d0 = (job & 3) * 16;
        const _Float16* ap = PRB + (s0 + lo) * 64 + (hi << 3);
        const _Float16* bp = PVT + (d0 + lo) * 64 + (hi << 3);
        v8f acc = {};
        #pragma unroll
        for (int kk = 0; kk < 64; kk += 32) {
            v16h a = load_frag(ap + kk, ap + kk + 16);
            v16h bb = load_frag(bp + kk, bp + kk + 16);
            acc = wmma_f16(a, bb, acc);
        }
        #pragma unroll
        for (int i = 0; i < 8; ++i) {
            int s = s0 + i + (hi << 3);
            if (s < SEQ)
                CTX[base + (size_t)s * HID + d0 + lo] = (_Float16)acc[i];
        }
    }
}

// ---------------- launcher ----------------

extern "C" void kernel_launch(void* const* d_in, const int* in_sizes, int n_in,
                              void* d_out, int out_size, void* d_ws, size_t ws_size,
                              hipStream_t stream) {
    const float* X  = (const float*)d_in[0];
    const float* Wq = (const float*)d_in[1];  const float* bq = (const float*)d_in[2];
    const float* Wk = (const float*)d_in[3];  const float* bk = (const float*)d_in[4];
    const float* Wv = (const float*)d_in[5];  const float* bv = (const float*)d_in[6];
    const float* Wo = (const float*)d_in[7];  const float* bo = (const float*)d_in[8];
    const float* Ek = (const float*)d_in[9];  const float* Ev = (const float*)d_in[10];

    char* ws = (char*)d_ws;
    const size_t SZ_ACT = (size_t)NTOK * HID * 2;   // 19,365,888 B
    const size_t SZ_W   = (size_t)HID * HID * 2;    //  1,179,648 B
    const size_t SZ_E   = (size_t)64 * SPAD * 2;    //     28,672 B
    _Float16* X16  = (_Float16*)(ws);
    _Float16* Q16  = (_Float16*)(ws + SZ_ACT);
    _Float16* K16  = (_Float16*)(ws + 2 * SZ_ACT);
    _Float16* V16  = (_Float16*)(ws + 3 * SZ_ACT);
    _Float16* C16  = (_Float16*)(ws + 4 * SZ_ACT);
    _Float16* WqT  = (_Float16*)(ws + 5 * SZ_ACT);
    _Float16* WkT  = (_Float16*)(ws + 5 * SZ_ACT + SZ_W);
    _Float16* WvT  = (_Float16*)(ws + 5 * SZ_ACT + 2 * SZ_W);
    _Float16* WoT  = (_Float16*)(ws + 5 * SZ_ACT + 3 * SZ_W);
    _Float16* EkT  = (_Float16*)(ws + 5 * SZ_ACT + 4 * SZ_W);
    _Float16* EvT  = (_Float16*)(ws + 5 * SZ_ACT + 4 * SZ_W + SZ_E);

    const int NELEM8 = NTOK * HID / 8;              // 1,210,368
    cvt_f32_to_f16_v8<<<(NELEM8 + 255) / 256, 256, 0, stream>>>(X, X16, NELEM8);
    transpose_w<<<144, 256, 0, stream>>>(Wq, WqT);
    transpose_w<<<144, 256, 0, stream>>>(Wk, WkT);
    transpose_w<<<144, 256, 0, stream>>>(Wv, WvT);
    transpose_w<<<144, 256, 0, stream>>>(Wo, WoT);
    transpose_e<<<(64 * SPAD + 255) / 256, 256, 0, stream>>>(Ek, EkT);
    transpose_e<<<(64 * SPAD + 255) / 256, 256, 0, stream>>>(Ev, EvT);

    const int MBLK = (MTILES + 15) / 16;            // 50
    const int GBLK = MBLK * 12;                     // 600 blocks (256M x 64N tiles)
    gemm_768<true><<<GBLK, 256, 0, stream>>>(X16, WqT, bq, (void*)Q16);
    gemm_768<true><<<GBLK, 256, 0, stream>>>(X16, WkT, bk, (void*)K16);
    gemm_768<true><<<GBLK, 256, 0, stream>>>(X16, WvT, bv, (void*)V16);

    lr_attn_core<<<64 * 12, 256, 0, stream>>>(Q16, K16, V16, EkT, EvT, C16);

    gemm_768<false><<<GBLK, 256, 0, stream>>>(C16, WoT, bo, d_out);
}
